// SparseMultiHeadAttention_14053132992808
// MI455X (gfx1250) — matrix operand order repaired and checked by reading, not compiled, not run on hardware
//
#include <hip/hip_runtime.h>

// ---------------- constants (hardcoded from reference) ----------------
#define BATCH   4
#define SEQ     2048
#define DMODEL  1024
#define NHEAD   16
#define DKH     64          // head dim
#define NQT     (SEQ / 16)  // 128 query tiles per sequence
#define WIN     256

typedef __attribute__((ext_vector_type(16))) __bf16 v16bf;
typedef __attribute__((ext_vector_type(8)))  __bf16 v8bf;
typedef __attribute__((ext_vector_type(8)))  float  v8f;

__device__ __forceinline__ v8f zero8() {
  v8f z;
#pragma unroll
  for (int i = 0; i < 8; ++i) z[i] = 0.0f;
  return z;
}

// ---------------------------------------------------------------------
// Elementwise fp32 -> bf16 (8 elements / thread, 32B load -> 16B store)
// ---------------------------------------------------------------------
__global__ __launch_bounds__(256) void cvt_bf16_k(const float* __restrict__ in,
                                                  __bf16* __restrict__ out) {
  size_t idx = ((size_t)blockIdx.x * 256 + threadIdx.x) * 8;
  v8f x = *(const v8f*)(in + idx);
  v8bf y;
#pragma unroll
  for (int i = 0; i < 8; ++i) y[i] = (__bf16)x[i];
  *(v8bf*)(out + idx) = y;
}

// ---------------------------------------------------------------------
// Transpose fp32 weight W[k][n] (1024x1024) -> bf16 Wt[n][k]
// ---------------------------------------------------------------------
__global__ __launch_bounds__(256) void transpose_w_bf16(
    const float* __restrict__ in, __bf16* __restrict__ out) {
  __shared__ float tile[32][33];
  int bx = blockIdx.x * 32, by = blockIdx.y * 32;
  int tx = threadIdx.x, ty = threadIdx.y;  // 32 x 8
#pragma unroll
  for (int i = 0; i < 32; i += 8)
    tile[ty + i][tx] = in[(size_t)(by + ty + i) * DMODEL + bx + tx];
  __syncthreads();
#pragma unroll
  for (int i = 0; i < 32; i += 8)
    out[(size_t)(bx + ty + i) * DMODEL + by + tx] = (__bf16)tile[tx][ty + i];
}

// ---------------------------------------------------------------------
// GEMM: C(8192x1024) = A_bf16(8192x1024) @ W(1024x1024) + bias
//   Wt is pre-transposed bf16 [n][k]; A pre-converted bf16 row-major.
//   Wave tile: 32(M) x 64(N); block = 8 waves -> 256(M) x 64(N).
// MODE 0: bf16 out, layout [b][h][s][dk]      (Q, K)
// MODE 1: bf16 out, layout [b][h][dk][s]      (V, transposed for PV WMMA)
// MODE 2: f32 out, row-major [m][n]           (final projection)
// ---------------------------------------------------------------------
template <int MODE>
__global__ __launch_bounds__(256) void gemm_bias_k(
    const __bf16* __restrict__ A, const __bf16* __restrict__ Wt,
    const float* __restrict__ bias, void* __restrict__ outp) {
  const int K = DMODEL;
  int lane  = threadIdx.x & 31;
  int wave  = threadIdx.x >> 5;
  int lhalf = lane >> 4;   // 0/1
  int lmod  = lane & 15;
  int mbase = blockIdx.y * 256 + wave * 32;
  int nbase = blockIdx.x * 64;

  v8f c[2][4];
#pragma unroll
  for (int mi = 0; mi < 2; ++mi)
#pragma unroll
    for (int ni = 0; ni < 4; ++ni) c[mi][ni] = zero8();

#pragma unroll 2
  for (int kc = 0; kc < K; kc += 32) {
    // A fragments: 16x32 bf16, two contiguous 16B chunks per lane
    v16bf a[2];
#pragma unroll
    for (int mi = 0; mi < 2; ++mi) {
      const __bf16* ap =
          A + (size_t)(mbase + mi * 16 + lmod) * K + kc + lhalf * 16;
      v8bf lo = *(const v8bf*)(ap);
      v8bf hi = *(const v8bf*)(ap + 8);
#pragma unroll
      for (int i = 0; i < 8; ++i) {
        a[mi][i]     = lo[i];
        a[mi][8 + i] = hi[i];
      }
    }
    // B fragments: 32x16 bf16 from Wt[n][k] (32B contiguous per lane)
    v16bf bfr[4];
#pragma unroll
    for (int ni = 0; ni < 4; ++ni)
      bfr[ni] = *(const v16bf*)(Wt + (size_t)(nbase + ni * 16 + lmod) * K +
                                kc + lhalf * 16);
#pragma unroll
    for (int mi = 0; mi < 2; ++mi)
#pragma unroll
      for (int ni = 0; ni < 4; ++ni)
        c[mi][ni] = __builtin_amdgcn_wmma_f32_16x16x32_bf16(
            false, a[mi], false, bfr[ni], (short)0, c[mi][ni], false, false);
  }

  // bias + store (C layout: row = r + 8*lhalf, col = lmod)
#pragma unroll
  for (int ni = 0; ni < 4; ++ni) {
    int cn = nbase + ni * 16 + lmod;
    float bv = bias[cn];
#pragma unroll
    for (int mi = 0; mi < 2; ++mi) {
#pragma unroll
      for (int r = 0; r < 8; ++r) {
        float val = c[mi][ni][r] + bv;
        int rm = mbase + mi * 16 + r + lhalf * 8;
        if (MODE == 2) {
          ((float*)outp)[(size_t)rm * DMODEL + cn] = val;
        } else {
          int b = rm >> 11, srow = rm & (SEQ - 1);
          int h = cn >> 6, dk = cn & (DKH - 1);
          size_t addr;
          if (MODE == 0)
            addr = ((size_t)(b * NHEAD + h) * SEQ + srow) * DKH + dk;
          else
            addr = ((size_t)(b * NHEAD + h) * DKH + dk) * SEQ + srow;
          ((__bf16*)outp)[addr] = (__bf16)val;
        }
      }
    }
  }
}

// ---------------------------------------------------------------------
// Windowed flash attention.
//   One wave = one (b,h,query-tile of 16). 32 keys per step (2 tiles).
//   Q,K bf16 [b][h][s][dk]; V bf16 [b][h][dk][s]; output bf16 [b][s][d].
// ---------------------------------------------------------------------
__global__ __launch_bounds__(256) void attn_k(
    const __bf16* __restrict__ Qp, const __bf16* __restrict__ Kp,
    const __bf16* __restrict__ Vp, const float* __restrict__ amask,
    __bf16* __restrict__ Ob) {
  __shared__ float pls[8][16][33];  // per-wave private P scratch

  int lane  = threadIdx.x & 31;
  int wave  = threadIdx.x >> 5;
  int lhalf = lane >> 4;
  int lmod  = lane & 15;
  int qt = blockIdx.x;                  // 0..127
  int bh = blockIdx.y * 8 + wave;       // 0..63
  int b  = bh >> 4;
  int h  = bh & 15;

  const __bf16* Qh = Qp + (size_t)bh * SEQ * DKH;
  const __bf16* Kh = Kp + (size_t)bh * SEQ * DKH;
  const __bf16* Vh = Vp + (size_t)bh * DKH * SEQ;

  // Q A-fragments for dk chunks [0,32) and [32,64)
  v16bf qa[2];
#pragma unroll
  for (int cs = 0; cs < 2; ++cs) {
    const __bf16* qp0 =
        Qh + (size_t)(qt * 16 + lmod) * DKH + cs * 32 + lhalf * 16;
    v8bf lo = *(const v8bf*)qp0;
    v8bf hi = *(const v8bf*)(qp0 + 8);
#pragma unroll
    for (int i = 0; i < 8; ++i) {
      qa[cs][i]     = lo[i];
      qa[cs][8 + i] = hi[i];
    }
  }

  v8f O[4];
#pragma unroll
  for (int ni = 0; ni < 4; ++ni) O[ni] = zero8();
  float mrow[8], lrow[8];
#pragma unroll
  for (int r = 0; r < 8; ++r) { mrow[r] = -3.0e38f; lrow[r] = 0.0f; }

  // key-tile range: window tiles [qt-16, qt+16] plus global tile 0;
  // query tile 0 (global queries) attends everything.
  int lo = (qt <= 16) ? 0 : (qt - 16);
  int hi = (qt == 0) ? (NQT - 1) : ((qt + 16 < NQT - 1) ? qt + 16 : NQT - 1);
  int lo_e = lo & ~1;
  int extra = (lo_e > 0) ? 1 : 0;  // prepend pair (0,1) for global keys
  int npairs = (hi - lo_e) / 2 + 1 + extra;
  const float scale = 0.125f;  // 1/sqrt(64)

  for (int p = 0; p < npairs; ++p) {
    int kt = (extra && p == 0) ? 0 : (lo_e + (p - extra) * 2);

    // ---- scores: two 16x16 tiles = Q(16x64) @ K_tile^T ----
    v8f svec[2];
#pragma unroll
    for (int hf = 0; hf < 2; ++hf) {
      int ktile = kt + hf;
      v16bf kb0 = *(const v16bf*)(Kh + (size_t)(ktile * 16 + lmod) * DKH +
                                  lhalf * 16);
      v16bf kb1 = *(const v16bf*)(Kh + (size_t)(ktile * 16 + lmod) * DKH +
                                  32 + lhalf * 16);
      v8f s = zero8();
      s = __builtin_amdgcn_wmma_f32_16x16x32_bf16(false, qa[0], false, kb0,
                                                  (short)0, s, false, false);
      s = __builtin_amdgcn_wmma_f32_16x16x32_bf16(false, qa[1], false, kb1,
                                                  (short)0, s, false, false);
      svec[hf] = s;
    }

    float am0 = amask[b * SEQ + kt * 16 + lmod];
    float am1 = amask[b * SEQ + (kt + 1) * 16 + lmod];

    // ---- mask + scale + row max ----
    float rmax[8];
#pragma unroll
    for (int r = 0; r < 8; ++r) {
      int irow = qt * 16 + r + lhalf * 8;
      float rm = -3.0e38f;
#pragma unroll
      for (int hf = 0; hf < 2; ++hf) {
        int j = (kt + hf) * 16 + lmod;
        int d = irow - j;
        bool allowed = (irow < 16) || (j < 16) || (d <= WIN && d >= -WIN);
        float sval = svec[hf][r] * scale + (hf == 0 ? am0 : am1);
        sval = allowed ? sval : -3.0e38f;
        svec[hf][r] = sval;
        rm = fmaxf(rm, sval);
      }
#pragma unroll
      for (int off = 1; off < 16; off <<= 1)
        rm = fmaxf(rm, __shfl_xor(rm, off, 32));
      rmax[r] = rm;
    }

    // ---- online softmax update ----
#pragma unroll
    for (int r = 0; r < 8; ++r) {
      float mnew = fmaxf(mrow[r], rmax[r]);
      float corr = __expf(mrow[r] - mnew);
      mrow[r] = mnew;
      float psum = 0.0f;
#pragma unroll
      for (int hf = 0; hf < 2; ++hf) {
        float sval = svec[hf][r];
        float pv = (sval > -1.0e38f) ? __expf(sval - mnew) : 0.0f;
        svec[hf][r] = pv;
        psum += pv;
      }
#pragma unroll
      for (int off = 1; off < 16; off <<= 1)
        psum += __shfl_xor(psum, off, 32);
      lrow[r] = lrow[r] * corr + psum;
#pragma unroll
      for (int ni = 0; ni < 4; ++ni) O[ni][r] *= corr;
    }

    // ---- P: C-layout -> A-layout via per-wave LDS (in-order per wave) ----
    float(*pw)[33] = pls[wave];
#pragma unroll
    for (int r = 0; r < 8; ++r) {
      int m = r + lhalf * 8;
      pw[m][lmod]      = svec[0][r];
      pw[m][16 + lmod] = svec[1][r];
    }
    asm volatile("" ::: "memory");
    v16bf pa;
#pragma unroll
    for (int i = 0; i < 16; ++i) {
      int kk = lhalf * 16 + i;
      pa[i] = (__bf16)pw[lmod][kk];
    }
    asm volatile("" ::: "memory");

    // ---- O += P(16x32) @ V(32x64) ----
#pragma unroll
    for (int ni = 0; ni < 4; ++ni) {
      v16bf vb = *(const v16bf*)(Vh + (size_t)(ni * 16 + lmod) * SEQ +
                                 kt * 16 + lhalf * 16);
      O[ni] = __builtin_amdgcn_wmma_f32_16x16x32_bf16(false, pa, false, vb,
                                                      (short)0, O[ni], false,
                                                      false);
    }
  }

  // ---- finalize: divide by l, store bf16 [b][s][d] ----
#pragma unroll
  for (int r = 0; r < 8; ++r) {
    float inv = 1.0f / lrow[r];
    int srow = qt * 16 + r + lhalf * 8;
#pragma unroll
    for (int ni = 0; ni < 4; ++ni) {
      Ob[((size_t)(b * SEQ + srow)) * DMODEL + h * DKH + ni * 16 + lmod] =
          (__bf16)(O[ni][r] * inv);
    }
  }
}

// ---------------------------------------------------------------------
extern "C" void kernel_launch(void* const* d_in, const int* in_sizes, int n_in,
                              void* d_out, int out_size, void* d_ws,
                              size_t ws_size, hipStream_t stream) {
  (void)in_sizes; (void)n_in; (void)out_size; (void)ws_size;
  const float* q  = (const float*)d_in[0];
  const float* k  = (const float*)d_in[1];
  const float* v  = (const float*)d_in[2];
  const float* am = (const float*)d_in[3];
  const float* wq = (const float*)d_in[4];
  const float* bq = (const float*)d_in[5];
  const float* wk = (const float*)d_in[6];
  const float* bk = (const float*)d_in[7];
  const float* wv = (const float*)d_in[8];
  const float* bv = (const float*)d_in[9];
  const float* wo = (const float*)d_in[10];
  const float* bo = (const float*)d_in[11];

  char* ws = (char*)d_ws;
  const size_t MB = 1024 * 1024;
  __bf16* wqt = (__bf16*)(ws + 0 * MB);    // 2 MB each
  __bf16* wkt = (__bf16*)(ws + 2 * MB);
  __bf16* wvt = (__bf16*)(ws + 4 * MB);
  __bf16* wot = (__bf16*)(ws + 6 * MB);
  __bf16* qb  = (__bf16*)(ws + 8 * MB);    // 16 MB each, bf16 inputs
  __bf16* kb  = (__bf16*)(ws + 24 * MB);
  __bf16* vb  = (__bf16*)(ws + 40 * MB);
  __bf16* Qp  = (__bf16*)(ws + 56 * MB);   // 16 MB each (bf16 [b][h][s][dk])
  __bf16* Kp  = (__bf16*)(ws + 72 * MB);
  __bf16* Vp  = (__bf16*)(ws + 88 * MB);   // bf16 [b][h][dk][s]
  // attention output aliases qb: qb is dead once the Q projection GEMM has
  // run, and attn_k (which writes Ab) is stream-ordered after it.
  __bf16* Ab  = (__bf16*)(ws + 8 * MB);    // 16 MB bf16 [b][s][d]

  const int NELEM = BATCH * SEQ * DMODEL;  // 8388608
  dim3 cg(NELEM / (256 * 8));              // 4096 blocks
  cvt_bf16_k<<<cg, 256, 0, stream>>>(q, qb);
  cvt_bf16_k<<<cg, 256, 0, stream>>>(k, kb);
  cvt_bf16_k<<<cg, 256, 0, stream>>>(v, vb);

  dim3 tb(32, 8), tg(32, 32);
  transpose_w_bf16<<<tg, tb, 0, stream>>>(wq, wqt);
  transpose_w_bf16<<<tg, tb, 0, stream>>>(wk, wkt);
  transpose_w_bf16<<<tg, tb, 0, stream>>>(wv, wvt);
  transpose_w_bf16<<<tg, tb, 0, stream>>>(wo, wot);

  dim3 gg(DMODEL / 64, (BATCH * SEQ) / 256);  // 16 x 32
  gemm_bias_k<0><<<gg, 256, 0, stream>>>(qb, wqt, bq, (void*)Qp);
  gemm_bias_k<0><<<gg, 256, 0, stream>>>(kb, wkt, bk, (void*)Kp);
  gemm_bias_k<1><<<gg, 256, 0, stream>>>(vb, wvt, bv, (void*)Vp);

  attn_k<<<dim3(NQT, (BATCH * NHEAD) / 8), 256, 0, stream>>>(Qp, Kp, Vp, am,
                                                             Ab);

  gemm_bias_k<2><<<gg, 256, 0, stream>>>(Ab, wot, bo, d_out);
}
